// GdnMixer_12360915878685
// MI455X (gfx1250) — matrix operand order repaired and checked by reading, not compiled, not run on hardware
//
#include <hip/hip_runtime.h>
#include <hip/hip_bf16.h>
#include <math.h>

// ---- problem constants (match reference) ----
static constexpr int B_   = 2;
static constexpr int S_   = 256;
static constexpr int H_   = 2048;
static constexpr int HK_  = 16;
static constexpr int HV_  = 32;
static constexpr int DK_  = 128;
static constexpr int DV_  = 128;
static constexpr int KC_  = 4;                // conv taps
static constexpr int KEY_ = HK_ * DK_;        // 2048
static constexpr int VAL_ = HV_ * DV_;        // 4096
static constexpr int CONV_ = 2 * KEY_ + VAL_; // 8192
static constexpr int BS_  = B_ * S_;          // 512
static constexpr float EPS_ = 1e-6f;

typedef __attribute__((ext_vector_type(16))) __bf16 v16bf;
typedef __attribute__((ext_vector_type(8)))  float  v8f;

// single-instruction packed f32->bf16x2 convert (dst.l = lo, dst.h = hi);
// mnemonic/operand form verified against round-2 compiler output.
__device__ __forceinline__ unsigned pack_bf2(float lo, float hi)
{
    unsigned r;
    asm("v_cvt_pk_bf16_f32 %0, %1, %2" : "=v"(r) : "v"(lo), "v"(hi));
    return r;
}

// ======================================================================
// Tiled bf16-WMMA GEMM:  C[M,N](f32) = A[M,K](f32) * W[K,N](f32)
// Requirements: M%128==0, N%128==0, K%32==0 (all call sites satisfy this).
// Block tile 128x128, K-step 32, 256 threads = 8 waves in a 4(M)x2(N)
// grid; each wave owns a 32x64 tile -> 8 x v_wmma_f32_16x16x32_bf16 per
// K-step. Global->reg->LDS software pipeline; LDS holds packed bf16
// K-pairs so fragment loads are ds_load_b128 / ds_load_b32.
// ======================================================================
#define GBM 128
#define GBN 128
#define GBK 32

__global__ __launch_bounds__(256)
void gemm_bf16_wmma(const float* __restrict__ A, const float* __restrict__ W,
                    float* __restrict__ C, int M, int N, int Kdim)
{
    // packed-pair tiles: word [m][kp] = (A[m][2kp], A[m][2kp+1])
    __shared__ unsigned Asp[GBM][GBK / 2 + 4];   // stride 20 dwords -> 16B-aligned quads
    __shared__ unsigned Bsp[GBK / 2][GBN + 1];   // stride 129 dwords

    const int tid  = threadIdx.x;
    const int lane = tid & 31;
    const int wave = tid >> 5;
    const int wm   = wave >> 1;           // 0..3 (32-row slab)
    const int wn   = wave & 1;            // 0..1 (64-col slab)
    const int mBase = blockIdx.y * GBM;
    const int nBase = blockIdx.x * GBN;
    const int half = lane >> 4;           // 0..1
    const int l16  = lane & 15;

    // staging coordinates
    const int arow = tid >> 1;            // 0..127
    const int acp  = (tid & 1) * 8;       // packed-col base (dwords), elem col = acp*2
    const int bkp  = tid >> 4;            // 0..15 packed K row
    const int bc0  = (tid & 15) * 8;      // col base

    const float* Aptr  = A + (long)(mBase + arow) * Kdim + acp * 2;
    const float* Bptr0 = W + (long)(2 * bkp) * N + nBase + bc0;   // row 2*bkp
    const float* Bptr1 = Bptr0 + N;                               // row 2*bkp+1

    const v8f vzero = {0.f,0.f,0.f,0.f,0.f,0.f,0.f,0.f};
    v8f acc[2][4];
    #pragma unroll
    for (int i = 0; i < 2; ++i)
        #pragma unroll
        for (int j = 0; j < 4; ++j) acc[i][j] = vzero;

    // ---- prologue fetch (tile 0): A 16 floats, B 2x8 floats ----
    float4 ra0  = *(const float4*)(Aptr);
    float4 ra1  = *(const float4*)(Aptr + 4);
    float4 ra2  = *(const float4*)(Aptr + 8);
    float4 ra3  = *(const float4*)(Aptr + 12);
    float4 rb00 = *(const float4*)(Bptr0);
    float4 rb01 = *(const float4*)(Bptr0 + 4);
    float4 rb10 = *(const float4*)(Bptr1);
    float4 rb11 = *(const float4*)(Bptr1 + 4);

    union FragU { unsigned u[8]; v16bf v; };

    const int nK = Kdim / GBK;
    for (int kt = 0; kt < nK; ++kt) {
        // ---- commit staged registers to LDS (packed bf16 K-pairs) ----
        Asp[arow][acp + 0] = pack_bf2(ra0.x, ra0.y);
        Asp[arow][acp + 1] = pack_bf2(ra0.z, ra0.w);
        Asp[arow][acp + 2] = pack_bf2(ra1.x, ra1.y);
        Asp[arow][acp + 3] = pack_bf2(ra1.z, ra1.w);
        Asp[arow][acp + 4] = pack_bf2(ra2.x, ra2.y);
        Asp[arow][acp + 5] = pack_bf2(ra2.z, ra2.w);
        Asp[arow][acp + 6] = pack_bf2(ra3.x, ra3.y);
        Asp[arow][acp + 7] = pack_bf2(ra3.z, ra3.w);

        Bsp[bkp][bc0 + 0] = pack_bf2(rb00.x, rb10.x);
        Bsp[bkp][bc0 + 1] = pack_bf2(rb00.y, rb10.y);
        Bsp[bkp][bc0 + 2] = pack_bf2(rb00.z, rb10.z);
        Bsp[bkp][bc0 + 3] = pack_bf2(rb00.w, rb10.w);
        Bsp[bkp][bc0 + 4] = pack_bf2(rb01.x, rb11.x);
        Bsp[bkp][bc0 + 5] = pack_bf2(rb01.y, rb11.y);
        Bsp[bkp][bc0 + 6] = pack_bf2(rb01.z, rb11.z);
        Bsp[bkp][bc0 + 7] = pack_bf2(rb01.w, rb11.w);
        __syncthreads();

        // ---- fetch next tile into registers (overlaps with WMMA below) ----
        if (kt + 1 < nK) {
            const float* An = Aptr + (long)(kt + 1) * GBK;
            const long  boff = (long)(kt + 1) * GBK * N;
            ra0  = *(const float4*)(An);
            ra1  = *(const float4*)(An + 4);
            ra2  = *(const float4*)(An + 8);
            ra3  = *(const float4*)(An + 12);
            rb00 = *(const float4*)(Bptr0 + boff);
            rb01 = *(const float4*)(Bptr0 + boff + 4);
            rb10 = *(const float4*)(Bptr1 + boff);
            rb11 = *(const float4*)(Bptr1 + boff + 4);
        }

        // ---- build fragments (ISA 7.12.2 wave32 layouts) ----
        FragU fa[2], fb[4];
        #pragma unroll
        for (int sm = 0; sm < 2; ++sm) {
            const int m = wm * 32 + sm * 16 + l16;
            const uint4 lo = *(const uint4*)&Asp[m][half * 4];       // kp = half*4 .. +3
            const uint4 hi = *(const uint4*)&Asp[m][8 + half * 4];   // kp = 8+half*4 .. +3
            fa[sm].u[0] = lo.x; fa[sm].u[1] = lo.y; fa[sm].u[2] = lo.z; fa[sm].u[3] = lo.w;
            fa[sm].u[4] = hi.x; fa[sm].u[5] = hi.y; fa[sm].u[6] = hi.z; fa[sm].u[7] = hi.w;
        }
        #pragma unroll
        for (int sn = 0; sn < 4; ++sn) {
            const int n = wn * 64 + sn * 16 + l16;
            #pragma unroll
            for (int p = 0; p < 8; ++p)
                fb[sn].u[p] = Bsp[(p < 4) ? (half * 4 + p) : (8 + half * 4 + (p - 4))][n];   // kp = half*4+p (p<4), 8+half*4+(p-4) (p>=4)
        }

        #pragma unroll
        for (int sm = 0; sm < 2; ++sm)
            #pragma unroll
            for (int sn = 0; sn < 4; ++sn)
                acc[sm][sn] = __builtin_amdgcn_wmma_f32_16x16x32_bf16(
                    false, fa[sm].v, false, fb[sn].v,
                    (short)0, acc[sm][sn], false, false);

        __syncthreads();
    }

    // ---- epilogue: C/D layout rows m = base + 8*half + r ----
    #pragma unroll
    for (int sm = 0; sm < 2; ++sm)
        #pragma unroll
        for (int sn = 0; sn < 4; ++sn) {
            const int n = nBase + wn * 64 + sn * 16 + l16;
            #pragma unroll
            for (int r = 0; r < 8; ++r) {
                const int m = mBase + wm * 32 + sm * 16 + half * 8 + r;
                C[(long)m * N + n] = acc[sm][sn][r];
            }
        }
}

// ======================================================================
// Fused tiny projections + gating:
//   b = hs @ W_b -> beta = sigmoid(b)
//   a = hs @ W_a -> g = -exp(A_log) * softplus(a + dt_bias)
// One 256-thread block per (b,s) row; hs row staged in LDS.
// ======================================================================
__global__ __launch_bounds__(256)
void ba_gating_k(const float* __restrict__ hs,
                 const float* __restrict__ W_b, const float* __restrict__ W_a,
                 const float* __restrict__ A_log, const float* __restrict__ dt_bias,
                 float* __restrict__ g, float* __restrict__ beta)
{
    __shared__ float row[H_];      // 8 KB
    __shared__ float red[256];
    const int tid = threadIdx.x;
    const int bs  = blockIdx.x;

    const float4* src = (const float4*)(hs + (long)bs * H_);
    #pragma unroll
    for (int i = 0; i < H_ / 4 / 256; ++i)
        ((float4*)row)[tid + i * 256] = src[tid + i * 256];
    __syncthreads();

    const int col = tid & 63;           // 0..31 -> b, 32..63 -> a
    const int h   = col & 31;
    const int seg = tid >> 6;           // 0..3
    const float* Wm = (col < 32) ? W_b : W_a;
    const int k0 = seg * (H_ / 4);

    float p = 0.f;
    #pragma unroll 8
    for (int k = 0; k < H_ / 4; ++k)
        p += row[k0 + k] * Wm[(long)(k0 + k) * HV_ + h];
    red[tid] = p;
    __syncthreads();

    if (tid < 64) {
        const float d = red[tid] + red[tid + 64] + red[tid + 128] + red[tid + 192];
        if (col < 32) {
            beta[bs * HV_ + h] = 1.f / (1.f + expf(-d));
        } else {
            const float x  = d + dt_bias[h];
            const float sp = (x > 20.f) ? x : log1pf(expf(x));
            g[bs * HV_ + h] = -expf(A_log[h]) * sp;
        }
    }
}

// ======================================================================
// conv_state_out[b][c][t] = mixed_raw[b][S-3+t][c]
// ======================================================================
__global__ void conv_state_out_k(const float* __restrict__ mixed, float* __restrict__ cso)
{
    const int idx = blockIdx.x * blockDim.x + threadIdx.x;
    const int total = B_ * CONV_ * (KC_ - 1);
    if (idx >= total) return;
    const int t = idx % (KC_ - 1);
    const int c = (idx / (KC_ - 1)) % CONV_;
    const int b = idx / ((KC_ - 1) * CONV_);
    cso[idx] = mixed[((long)b * S_ + (S_ - (KC_ - 1) + t)) * CONV_ + c];
}

// ======================================================================
// causal depthwise conv1d (K=4, stateful) + SiLU
// ======================================================================
__global__ void conv_silu_k(const float* __restrict__ mixed,
                            const float* __restrict__ conv_state,
                            const float* __restrict__ conv_w,
                            const float* __restrict__ conv_b,
                            float* __restrict__ mconv)
{
    const long idx = (long)blockIdx.x * blockDim.x + threadIdx.x;
    const long total = (long)BS_ * CONV_;
    if (idx >= total) return;
    const int c  = (int)(idx % CONV_);
    const int bs = (int)(idx / CONV_);
    const int s  = bs % S_;
    const int b  = bs / S_;

    float acc = conv_b[c];
    #pragma unroll
    for (int j = 0; j < KC_; ++j) {
        const int t = s + j;
        const float x = (t < KC_ - 1)
            ? conv_state[((long)b * CONV_ + c) * (KC_ - 1) + t]
            : mixed[((long)b * S_ + (t - (KC_ - 1))) * CONV_ + c];
        acc += x * conv_w[c * KC_ + j];
    }
    mconv[idx] = acc * (1.f / (1.f + expf(-acc)));   // SiLU
}

// ======================================================================
// per-head L2 norm of q (scaled DK^-0.5) and k; 128 threads per (b,s,hk)
// ======================================================================
__global__ __launch_bounds__(128)
void qknorm_k(const float* __restrict__ mconv,
              float* __restrict__ qn, float* __restrict__ kn)
{
    __shared__ float red[128];
    const int tid = threadIdx.x;
    const int grp = blockIdx.x;            // bs*HK + hk
    const int hk  = grp % HK_;
    const int bs  = grp / HK_;

    const float qv = mconv[(long)bs * CONV_ + hk * DK_ + tid];
    const float kv = mconv[(long)bs * CONV_ + KEY_ + hk * DK_ + tid];

    red[tid] = qv * qv; __syncthreads();
    #pragma unroll
    for (int st = 64; st > 0; st >>= 1) { if (tid < st) red[tid] += red[tid + st]; __syncthreads(); }
    const float qscale = rsqrtf(red[0] + EPS_) * 0.08838834764831845f;  // DK^-0.5
    __syncthreads();

    red[tid] = kv * kv; __syncthreads();
    #pragma unroll
    for (int st = 64; st > 0; st >>= 1) { if (tid < st) red[tid] += red[tid + st]; __syncthreads(); }
    const float kscale = rsqrtf(red[0] + EPS_);

    qn[(long)grp * DK_ + tid] = qv * qscale;
    kn[(long)grp * DK_ + tid] = kv * kscale;
}

// ======================================================================
// gated delta-rule recurrent scan. One 128-thread block per (b, head_v);
// lane owns one state column (128 f32 in VGPRs); k/q via LDS broadcast.
// ======================================================================
__global__ __launch_bounds__(128)
void scan_k(const float* __restrict__ qn, const float* __restrict__ kn,
            const float* __restrict__ mconv,
            const float* __restrict__ g, const float* __restrict__ beta,
            const int* __restrict__ ctx, const float* __restrict__ S0,
            float* __restrict__ o, float* __restrict__ Sfin)
{
    __shared__ float ksh[DK_];
    __shared__ float qsh[DK_];

    const int tid = threadIdx.x;            // v index (state column)
    const int bh  = blockIdx.x;             // b*HV + h
    const int h   = bh % HV_;
    const int b   = bh / HV_;
    const int hk  = h >> 1;                 // rep = HV/HK = 2
    const int ctxb = ctx[b];

    float Sreg[DK_];
    #pragma unroll
    for (int kx = 0; kx < DK_; ++kx)
        Sreg[kx] = S0[((long)bh * DK_ + kx) * DV_ + tid];

    for (int s = 0; s < S_; ++s) {
        const long bsrow = (long)b * S_ + s;
        ksh[tid] = kn[(bsrow * HK_ + hk) * DK_ + tid];
        qsh[tid] = qn[(bsrow * HK_ + hk) * DK_ + tid];
        __syncthreads();

        const float vv = mconv[bsrow * CONV_ + 2 * KEY_ + h * DV_ + tid];
        const float gt = g[bsrow * HV_ + h];
        const float bt = beta[bsrow * HV_ + h];

        float ov = 0.f;
        if (s < ctxb) {                      // wave-uniform branch
            const float eg = expf(gt);
            float kv0 = 0.f, kv1 = 0.f, kv2 = 0.f, kv3 = 0.f;
            #pragma unroll
            for (int kx = 0; kx < DK_; kx += 4) {
                const float s0 = Sreg[kx]     * eg;
                const float s1 = Sreg[kx + 1] * eg;
                const float s2 = Sreg[kx + 2] * eg;
                const float s3 = Sreg[kx + 3] * eg;
                kv0 += ksh[kx] * s0;     kv1 += ksh[kx + 1] * s1;
                kv2 += ksh[kx + 2] * s2; kv3 += ksh[kx + 3] * s3;
                Sreg[kx] = s0; Sreg[kx + 1] = s1; Sreg[kx + 2] = s2; Sreg[kx + 3] = s3;
            }
            const float delta = (vv - (kv0 + kv1 + kv2 + kv3)) * bt;

            float o0 = 0.f, o1 = 0.f, o2 = 0.f, o3 = 0.f;
            #pragma unroll
            for (int kx = 0; kx < DK_; kx += 4) {
                const float s0 = Sreg[kx]     + ksh[kx]     * delta;
                const float s1 = Sreg[kx + 1] + ksh[kx + 1] * delta;
                const float s2 = Sreg[kx + 2] + ksh[kx + 2] * delta;
                const float s3 = Sreg[kx + 3] + ksh[kx + 3] * delta;
                o0 += qsh[kx] * s0;      o1 += qsh[kx + 1] * s1;
                o2 += qsh[kx + 2] * s2;  o3 += qsh[kx + 3] * s3;
                Sreg[kx] = s0; Sreg[kx + 1] = s1; Sreg[kx + 2] = s2; Sreg[kx + 3] = s3;
            }
            ov = o0 + o1 + o2 + o3;
        }
        o[(bsrow * HV_ + h) * DV_ + tid] = ov;
        __syncthreads();
    }

    #pragma unroll
    for (int kx = 0; kx < DK_; ++kx)
        Sfin[((long)bh * DK_ + kx) * DV_ + tid] = Sreg[kx];
}

// ======================================================================
// gated RMSNorm over DV + SiLU(z) gate
// ======================================================================
__global__ __launch_bounds__(128)
void rmsgate_k(const float* __restrict__ o, const float* __restrict__ z,
               const float* __restrict__ norm_w, float* __restrict__ gated)
{
    __shared__ float red[128];
    const int tid = threadIdx.x;
    const int grp = blockIdx.x;            // bs*HV + h
    const int h   = grp % HV_;
    const int bs  = grp / HV_;

    const float ov = o[(long)grp * DV_ + tid];
    red[tid] = ov * ov; __syncthreads();
    #pragma unroll
    for (int st = 64; st > 0; st >>= 1) { if (tid < st) red[tid] += red[tid + st]; __syncthreads(); }
    const float scale = rsqrtf(red[0] * (1.f / DV_) + EPS_);

    const float zv = z[(long)bs * VAL_ + h * DV_ + tid];
    const float sz = zv * (1.f / (1.f + expf(-zv)));
    gated[(long)grp * DV_ + tid] = ov * scale * norm_w[tid] * sz;
}

// ======================================================================
// launch
// ======================================================================
extern "C" void kernel_launch(void* const* d_in, const int* in_sizes, int n_in,
                              void* d_out, int out_size, void* d_ws, size_t ws_size,
                              hipStream_t stream)
{
    (void)in_sizes; (void)n_in; (void)out_size; (void)ws_size;

    const float* hs        = (const float*)d_in[0];
    const float* convstate = (const float*)d_in[1];
    const float* recstate  = (const float*)d_in[2];
    const int*   ctx       = (const int*)  d_in[3];
    const float* W_qkv     = (const float*)d_in[4];
    const float* W_z       = (const float*)d_in[5];
    const float* W_b       = (const float*)d_in[6];
    const float* W_a       = (const float*)d_in[7];
    const float* conv_w    = (const float*)d_in[8];
    const float* conv_b    = (const float*)d_in[9];
    const float* A_log     = (const float*)d_in[10];
    const float* dt_bias   = (const float*)d_in[11];
    const float* norm_w    = (const float*)d_in[12];
    const float* W_out     = (const float*)d_in[13];

    float* out = (float*)d_out;
    float* out_main = out;                                        // [B,S,H]
    float* out_cso  = out + (size_t)BS_ * H_;                     // [B,CONV,3]
    float* out_Sfin = out_cso + (size_t)B_ * CONV_ * (KC_ - 1);   // [B,HV,DK,DV]

    // workspace carve-up (floats)
    float* ws    = (float*)d_ws;
    float* mixed = ws;                                     // BS*CONV
    float* zbuf  = mixed + (size_t)BS_ * CONV_;            // BS*VAL
    float* mconv = zbuf  + (size_t)BS_ * VAL_;             // BS*CONV
    float* qnbuf = mconv + (size_t)BS_ * CONV_;            // BS*HK*DK
    float* knbuf = qnbuf + (size_t)BS_ * HK_ * DK_;        // BS*HK*DK
    float* gbuf  = knbuf + (size_t)BS_ * HK_ * DK_;        // BS*HV
    float* bebuf = gbuf  + (size_t)BS_ * HV_;              // BS*HV
    float* obuf  = bebuf + (size_t)BS_ * HV_;              // BS*HV*DV
    float* gated = qnbuf;  // overlay: qn/kn dead after scan; 2*BS*HK*DK == BS*VAL

    // 1-2) big input projections (bf16 WMMA GEMMs, guard-free)
    gemm_bf16_wmma<<<dim3(CONV_ / GBN, BS_ / GBM), 256, 0, stream>>>(hs, W_qkv, mixed, BS_, CONV_, H_);
    gemm_bf16_wmma<<<dim3(VAL_  / GBN, BS_ / GBM), 256, 0, stream>>>(hs, W_z,   zbuf,  BS_, VAL_,  H_);

    // 3) tiny b/a projections fused with gating math
    ba_gating_k<<<BS_, 256, 0, stream>>>(hs, W_b, W_a, A_log, dt_bias, gbuf, bebuf);

    // 4) conv state output (raw last K-1 inputs)
    {
        const int total = B_ * CONV_ * (KC_ - 1);
        conv_state_out_k<<<(total + 255) / 256, 256, 0, stream>>>(mixed, out_cso);
    }
    // 5) causal depthwise conv + SiLU
    {
        const long total = (long)BS_ * CONV_;
        conv_silu_k<<<(int)((total + 255) / 256), 256, 0, stream>>>(mixed, convstate, conv_w, conv_b, mconv);
    }
    // 6) q/k L2 norm
    qknorm_k<<<BS_ * HK_, 128, 0, stream>>>(mconv, qnbuf, knbuf);

    // 7) recurrent delta-rule scan (state in VGPRs)
    scan_k<<<B_ * HV_, 128, 0, stream>>>(qnbuf, knbuf, mconv, gbuf, bebuf, ctx, recstate, obuf, out_Sfin);

    // 8) gated RMSNorm * SiLU(z)
    rmsgate_k<<<BS_ * HV_, 128, 0, stream>>>(obuf, zbuf, norm_w, gated);

    // 9) output projection
    gemm_bf16_wmma<<<dim3(H_ / GBN, BS_ / GBM), 256, 0, stream>>>(gated, W_out, out_main, BS_, H_, VAL_);
}
